// CausalSelfAttention_65644280152230
// MI455X (gfx1250) — matrix operand order repaired and checked by reading, not compiled, not run on hardware
//
#include <hip/hip_runtime.h>
#include <hip/hip_bf16.h>
#include <math.h>

typedef __attribute__((ext_vector_type(16))) _Float16 v16h;
typedef __attribute__((ext_vector_type(8)))  _Float16 v8h;
typedef __attribute__((ext_vector_type(4)))  _Float16 v4h;
typedef __attribute__((ext_vector_type(8)))  float    v8f;

#define WMMA_F16(A, B, C) \
  __builtin_amdgcn_wmma_f32_16x16x32_f16(false, (A), false, (B), (short)0, (C), false, false)

__device__ __forceinline__ v16h mk16(v8h lo, v8h hi) {
  return __builtin_shufflevector(lo, hi, 0, 1, 2, 3, 4, 5, 6, 7, 8, 9, 10, 11, 12, 13, 14, 15);
}

static constexpr int BB = 2;            // batch
static constexpr int S  = 2048;         // sequence
static constexpr int D  = 1024;         // d_model
static constexpr int H  = 16;           // heads
static constexpr int Dh = 64;           // head dim

// Q pre-scale: 1/sqrt(Dh) * log2(e)  (softmax done in base-2: v_exp_f32 is 2^x)
static constexpr float QSCALE = 0.125f * 1.4426950408889634f;

// ---------------------------------------------------------------------------
// f32 -> f16 conversion (x), 4 elements per thread
// ---------------------------------------------------------------------------
__global__ void cvt_f32_f16(const float* __restrict__ src, _Float16* __restrict__ dst) {
  const int i = (blockIdx.x * blockDim.x + threadIdx.x) * 4;
  float4 v = *(const float4*)(src + i);
  v4h o;
  o[0] = (_Float16)v.x; o[1] = (_Float16)v.y; o[2] = (_Float16)v.z; o[3] = (_Float16)v.w;
  *(v4h*)(dst + i) = o;
}

// ---------------------------------------------------------------------------
// weight f32 [K][N] -> f16 transposed [N][K]   (K = N = 1024)
// ---------------------------------------------------------------------------
__global__ void cvt_transpose(const float* __restrict__ w, _Float16* __restrict__ wT) {
  const int i = blockIdx.x * blockDim.x + threadIdx.x;  // 1M threads
  const int n = i & (D - 1);
  const int k = i >> 10;
  wT[(size_t)n * D + k] = (_Float16)w[(size_t)k * D + n];
}

// ---------------------------------------------------------------------------
// GEMM: C[32x64 per wave] = A[M,K]f16 * BT[N,K]f16 + bias, K=D=1024
// Two A row-blocks share each B tile: 8 WMMAs per 12 B128 loads per 32-K step.
// Software-pipelined: K stepped by 64 with ping-pong operand buffers.
// mode 0: Q -> f16 [B,H,S,Dh], scaled by QSCALE
// mode 1: K -> f16 [B,H,S,Dh]
// mode 2: V -> f16 [B,H,Dh,S]  (transposed)
// mode 3: OUT -> f32 [M,N]
// ---------------------------------------------------------------------------
__global__ void gemm_wmma(const _Float16* __restrict__ A,
                          const _Float16* __restrict__ BT,
                          const float* __restrict__ bias,
                          void* __restrict__ dst, int mode) {
  const int lane = threadIdx.x & 31;
  const int wid  = (blockIdx.x * blockDim.x + threadIdx.x) >> 5;  // 0..2047
  const int g = lane >> 4, ln = lane & 15;
  const int m0 = (wid >> 4) * 32;
  const int n0 = (wid & 15) * 64;

  v8f zero = {};
  v8f c[2][4] = {{zero, zero, zero, zero}, {zero, zero, zero, zero}};

  // A chunk bases (two row-blocks): halves 0..15 at K+16g (contiguous)
  const _Float16* arow[2];
#pragma unroll
  for (int u = 0; u < 2; ++u) arow[u] = A + (size_t)(m0 + u * 16 + ln) * D + 16 * g;
  // B row base per tile: halves 0..15 at K+16g (contiguous)
  const _Float16* brow[4];
#pragma unroll
  for (int t = 0; t < 4; ++t) brow[t] = BT + (size_t)(n0 + t * 16 + ln) * D + 16 * g;

  // ---- prologue: load K-chunk 0 into buffer P
  v8h pa0[2], pa1[2], pb0[4], pb1[4];
#pragma unroll
  for (int u = 0; u < 2; ++u) {
    pa0[u] = *(const v8h*)(arow[u]);
    pa1[u] = *(const v8h*)(arow[u] + 8);
  }
#pragma unroll
  for (int t = 0; t < 4; ++t) {
    pb0[t] = *(const v8h*)(brow[t]);
    pb1[t] = *(const v8h*)(brow[t] + 8);
  }

  for (int k0 = 0; k0 < D; k0 += 64) {
    const int k1 = k0 + 32;
    const int k2 = (k0 + 64) & (D - 1);   // wrap on last iter (harmless refetch, no branch)

    // ---- issue loads for k1 into buffer Q
    v8h qa0[2], qa1[2], qb0[4], qb1[4];
#pragma unroll
    for (int u = 0; u < 2; ++u) {
      qa0[u] = *(const v8h*)(arow[u] + k1);
      qa1[u] = *(const v8h*)(arow[u] + k1 + 8);
    }
#pragma unroll
    for (int t = 0; t < 4; ++t) {
      qb0[t] = *(const v8h*)(brow[t] + k1);
      qb1[t] = *(const v8h*)(brow[t] + k1 + 8);
    }

    // ---- compute with buffer P (k0)
#pragma unroll
    for (int u = 0; u < 2; ++u) {
      v16h a = mk16(pa0[u], pa1[u]);
#pragma unroll
      for (int t = 0; t < 4; ++t) c[u][t] = WMMA_F16(a, mk16(pb0[t], pb1[t]), c[u][t]);
    }

    // ---- issue loads for k2 into buffer P
#pragma unroll
    for (int u = 0; u < 2; ++u) {
      pa0[u] = *(const v8h*)(arow[u] + k2);
      pa1[u] = *(const v8h*)(arow[u] + k2 + 8);
    }
#pragma unroll
    for (int t = 0; t < 4; ++t) {
      pb0[t] = *(const v8h*)(brow[t] + k2);
      pb1[t] = *(const v8h*)(brow[t] + k2 + 8);
    }

    // ---- compute with buffer Q (k1)
#pragma unroll
    for (int u = 0; u < 2; ++u) {
      v16h a = mk16(qa0[u], qa1[u]);
#pragma unroll
      for (int t = 0; t < 4; ++t) c[u][t] = WMMA_F16(a, mk16(qb0[t], qb1[t]), c[u][t]);
    }
  }

#pragma unroll
  for (int u = 0; u < 2; ++u) {
#pragma unroll
    for (int t = 0; t < 4; ++t) {
      const int n = n0 + t * 16 + ln;
      const float bval = bias[n];
#pragma unroll
      for (int r = 0; r < 8; ++r) {
        const int mr = m0 + u * 16 + r + 8 * g;    // row in [0,4096)
        float val = c[u][t][r] + bval;
        if (mode == 3) {
          ((float*)dst)[(size_t)mr * D + n] = val;
        } else {
          const int bb = mr >> 11, s = mr & (S - 1);
          const int hh = n >> 6, d = n & (Dh - 1);
          if (mode == 2) {
            ((_Float16*)dst)[(((size_t)(bb * H + hh)) * Dh + d) * S + s] = (_Float16)val;
          } else {
            float vv = (mode == 0) ? val * QSCALE : val;
            ((_Float16*)dst)[(((size_t)(bb * H + hh)) * S + s) * Dh + d] = (_Float16)vv;
          }
        }
      }
    }
  }
}

// ---------------------------------------------------------------------------
// Flash-style causal attention. One wave per 16-query block.
// K tiles for iter i+1 and V tiles for iter i are issued before the softmax
// VALU block + LDS bounce, so their latency is covered.
// Softmax is in base-2 (Q pre-scaled by log2e/sqrt(Dh)); exponentials are
// bare v_exp_f32.
// qh: [B,H,S,Dh] f16 (pre-scaled), kh: [B,H,S,Dh] f16, vt: [B,H,Dh,S] f16
// outh: [B,S,H*Dh] f16
// ---------------------------------------------------------------------------
__global__ void attn_wmma(const _Float16* __restrict__ qh,
                          const _Float16* __restrict__ kh,
                          const _Float16* __restrict__ vt,
                          _Float16* __restrict__ outh) {
  __shared__ _Float16 lds[4][16 * 40];  // per-wave P staging, stride 40 halves (16B aligned)

  const int lane  = threadIdx.x & 31;
  const int wslot = threadIdx.x >> 5;
  const int wid   = blockIdx.x * 4 + wslot;          // 0..4095
  const int qb = wid & 127;
  const int hh = (wid >> 7) & 15;
  const int b  = wid >> 11;
  const int g = lane >> 4, ln = lane & 15;
  const int q0 = qb * 16;
  const size_t head = (size_t)(b * H + hh) * S * Dh; // same offset for [S,Dh] and [Dh,S]

  // Q as A-operand, two K-chunks over Dh=64
  v16h aQ[2];
  {
    const _Float16* qrow = qh + head + (size_t)(q0 + ln) * Dh;
#pragma unroll
    for (int cc = 0; cc < 2; ++cc)
      aQ[cc] = mk16(*(const v8h*)(qrow + cc * 32 + 16 * g),
                    *(const v8h*)(qrow + cc * 32 + 8 + 16 * g));
  }

  // per-tile row bases (advance by k0*Dh / k0 per iteration)
  const _Float16* kbase[2];
  const _Float16* vbase[4];
#pragma unroll
  for (int t = 0; t < 2; ++t) kbase[t] = kh + head + (size_t)(t * 16 + ln) * Dh + 16 * g;
#pragma unroll
  for (int t = 0; t < 4; ++t) vbase[t] = vt + head + (size_t)(t * 16 + ln) * S + 16 * g;

  v8f zero = {};
  v8f o[4] = {zero, zero, zero, zero};
  float mrow[8], lrow[8];
#pragma unroll
  for (int r = 0; r < 8; ++r) { mrow[r] = -INFINITY; lrow[r] = 0.f; }

  const int nk = (q0 + 16 + 31) >> 5;  // 32-key iterations up to diagonal

  // ---- preload K tiles for iteration 0: kb[tile][dchunk][lo/hi]
  v8h kbuf[2][2][2];
#pragma unroll
  for (int t = 0; t < 2; ++t) {
    kbuf[t][0][0] = *(const v8h*)(kbase[t]);
    kbuf[t][0][1] = *(const v8h*)(kbase[t] + 8);
    kbuf[t][1][0] = *(const v8h*)(kbase[t] + 32);
    kbuf[t][1][1] = *(const v8h*)(kbase[t] + 40);
  }

  for (int it = 0; it < nk; ++it) {
    const int k0 = it * 32;

    // ---- scores: 16 queries x 32 keys, contraction over Dh=64 (preloaded K)
    v8f sc[2];
#pragma unroll
    for (int t = 0; t < 2; ++t) {
      v8f acc = zero;
      acc = WMMA_F16(aQ[0], mk16(kbuf[t][0][0], kbuf[t][0][1]), acc);
      acc = WMMA_F16(aQ[1], mk16(kbuf[t][1][0], kbuf[t][1][1]), acc);
      sc[t] = acc;
    }

    // ---- issue V loads for this iteration (consumed after softmax)
    v8h vbuf[4][2];
#pragma unroll
    for (int t = 0; t < 4; ++t) {
      vbuf[t][0] = *(const v8h*)(vbase[t] + k0);
      vbuf[t][1] = *(const v8h*)(vbase[t] + k0 + 8);
    }

    // ---- issue K loads for next iteration (wrap to 0 on last iter, no branch)
    {
      const size_t koff = (size_t)((k0 + 32) & (S - 1)) * Dh;
#pragma unroll
      for (int t = 0; t < 2; ++t) {
        kbuf[t][0][0] = *(const v8h*)(kbase[t] + koff);
        kbuf[t][0][1] = *(const v8h*)(kbase[t] + koff + 8);
        kbuf[t][1][0] = *(const v8h*)(kbase[t] + koff + 32);
        kbuf[t][1][1] = *(const v8h*)(kbase[t] + koff + 40);
      }
    }

    // ---- causal mask (only the diagonal-overlapping iteration)
    if (k0 + 31 > q0) {
#pragma unroll
      for (int t = 0; t < 2; ++t) {
        const int key = k0 + t * 16 + ln;
#pragma unroll
        for (int r = 0; r < 8; ++r) {
          const int qr = q0 + r + 8 * g;
          sc[t][r] = (key > qr) ? -INFINITY : sc[t][r];
        }
      }
    }

    // ---- online softmax, base-2 domain (rows: r + 8g per half-wave)
    v8f p0 = zero, p1 = zero;
    float alpha[8];
#pragma unroll
    for (int r = 0; r < 8; ++r) {
      float mx = fmaxf(sc[0][r], sc[1][r]);
#pragma unroll
      for (int dd = 1; dd < 16; dd <<= 1) mx = fmaxf(mx, __shfl_xor(mx, dd, 32));
      const float mnew = fmaxf(mrow[r], mx);
      const float e0 = __builtin_amdgcn_exp2f(sc[0][r] - mnew);
      const float e1 = __builtin_amdgcn_exp2f(sc[1][r] - mnew);
      p0[r] = e0; p1[r] = e1;
      float sum = e0 + e1;
#pragma unroll
      for (int dd = 1; dd < 16; dd <<= 1) sum += __shfl_xor(sum, dd, 32);
      const float a = __builtin_amdgcn_exp2f(mrow[r] - mnew);
      alpha[r] = a;
      lrow[r] = lrow[r] * a + sum;
      mrow[r] = mnew;
    }
#pragma unroll
    for (int t = 0; t < 4; ++t)
#pragma unroll
      for (int r = 0; r < 8; ++r) o[t][r] *= alpha[r];

    // ---- P (C-layout f32) -> LDS -> A-operand (f16) transpose bounce
    _Float16* L = lds[wslot];
#pragma unroll
    for (int r = 0; r < 8; ++r) {
      L[(r + 8 * g) * 40 + ln]      = (_Float16)p0[r];
      L[(r + 8 * g) * 40 + 16 + ln] = (_Float16)p1[r];
    }
    asm volatile("s_wait_dscnt 0" ::: "memory");   // per-wave DS ordering + compiler barrier
    v16h aP = mk16(*(const v8h*)&L[ln * 40 + 16 * g],
                   *(const v8h*)&L[ln * 40 + 8 + 16 * g]);
    asm volatile("" ::: "memory");

    // ---- O += P @ V  (contraction over 32 keys, 4 d-tiles, preloaded V)
#pragma unroll
    for (int t = 0; t < 4; ++t)
      o[t] = WMMA_F16(aP, mk16(vbuf[t][0], vbuf[t][1]), o[t]);
  }

  // ---- normalize and scatter to [B,S,H*Dh] f16
#pragma unroll
  for (int t = 0; t < 4; ++t) {
#pragma unroll
    for (int r = 0; r < 8; ++r) {
      const int qr = q0 + r + 8 * g;
      const float val = o[t][r] / lrow[r];
      outh[((size_t)(b * S + qr)) * D + hh * Dh + t * 16 + ln] = (_Float16)val;
    }
  }
}

// ---------------------------------------------------------------------------
extern "C" void kernel_launch(void* const* d_in, const int* in_sizes, int n_in,
                              void* d_out, int out_size, void* d_ws, size_t ws_size,
                              hipStream_t stream) {
  const float* x  = (const float*)d_in[0];
  const float* wq = (const float*)d_in[1];
  const float* bq = (const float*)d_in[2];
  const float* wk = (const float*)d_in[3];
  const float* bk = (const float*)d_in[4];
  const float* wv = (const float*)d_in[5];
  const float* bv = (const float*)d_in[6];
  const float* wo = (const float*)d_in[7];
  const float* bo = (const float*)d_in[8];
  float* out = (float*)d_out;

  char* ws = (char*)d_ws;
  _Float16* xh  = (_Float16*)(ws);                       //  8 MB  [M,D]
  _Float16* wqT = (_Float16*)(ws + (8u  << 20));         //  2 MB  [N,K]
  _Float16* wkT = (_Float16*)(ws + (10u << 20));
  _Float16* wvT = (_Float16*)(ws + (12u << 20));
  _Float16* woT = (_Float16*)(ws + (14u << 20));
  _Float16* qh  = (_Float16*)(ws + (16u << 20));         //  8 MB  [B,H,S,Dh]
  _Float16* kh  = (_Float16*)(ws + (24u << 20));         //  8 MB  [B,H,S,Dh]
  _Float16* vt  = (_Float16*)(ws + (32u << 20));         //  8 MB  [B,H,Dh,S]
  _Float16* ao  = (_Float16*)(ws + (40u << 20));         //  8 MB  [M,D]

  // 1) precision conversion (all f16 working set ~50MB -> resident in 192MB L2)
  cvt_f32_f16 <<<4096, 256, 0, stream>>>(x, xh);          // 4M elems / 4 per thread
  cvt_transpose<<<4096, 256, 0, stream>>>(wq, wqT);
  cvt_transpose<<<4096, 256, 0, stream>>>(wk, wkT);
  cvt_transpose<<<4096, 256, 0, stream>>>(wv, wvT);
  cvt_transpose<<<4096, 256, 0, stream>>>(wo, woT);

  // 2) QKV projections: 2048 waves each (exact fit, EXEC all-ones for WMMA)
  gemm_wmma<<<256, 256, 0, stream>>>(xh, wqT, bq, qh, 0);
  gemm_wmma<<<256, 256, 0, stream>>>(xh, wkT, bk, kh, 1);
  gemm_wmma<<<256, 256, 0, stream>>>(xh, wvT, bv, vt, 2);

  // 3) causal flash attention: one wave per 16-query block (4096 waves)
  attn_wmma<<<1024, 128, 0, stream>>>(qh, kh, vt, ao);

  // 4) output projection -> f32
  gemm_wmma<<<256, 256, 0, stream>>>(ao, woT, bo, out, 3);
}
